// ScanAttention_59047210385815
// MI455X (gfx1250) — matrix operand order repaired and checked by reading, not compiled, not run on hardware
//
#include <hip/hip_runtime.h>

typedef __attribute__((ext_vector_type(16))) _Float16 v16h;
typedef __attribute__((ext_vector_type(8)))  _Float16 v8h;
typedef __attribute__((ext_vector_type(8)))  float    v8f;

#define BATCH 2
#define QLEN  2048
#define KLEN  2048
#define HEADS 16
#define DHEAD 64

#define QTILE 64      // Q rows per block (4 waves x 16)
#define KTILE 64      // keys per iteration
#define ROWP  72      // padded f16 row length (144B, 16B aligned)

// ---- dynamic LDS layout (byte offsets; dynamic LDS base == 0, no static LDS) ----
#define STAGE_BUF   32768u                 // 16KB K f32 + 16KB V f32
#define STAGE_VOFF  16384u
#define KOFF        (2u * STAGE_BUF)              // 65536: K f16 [64][ROWP]
#define VOFF        (KOFF + 64u * ROWP * 2u)      // 74752: V f16 [d][ROWP] transposed
#define POFF        (VOFF + 64u * ROWP * 2u)      // 83968: P f16 [4][16][ROWP]
#define SMEM_TOTAL  (POFF + 4u * 16u * ROWP * 2u) // 93184 bytes

union V16U { v16h v; v8h h[2]; };

static __device__ __forceinline__ v8f wmma_f16(const v16h& a, const v16h& b, v8f c) {
    return __builtin_amdgcn_wmma_f32_16x16x32_f16(false, a, false, b, (short)0, c, false, false);
}

// Issue this thread's share of one K/V tile as async global->LDS b128 copies.
// Each thread: 8 x b128 for K + 8 x b128 for V (128 threads cover 64x64 f32 each).
static __device__ __forceinline__ void issue_tile_async(const float* kg, const float* vg,
                                                        unsigned lbase) {
    const uint64_t ka = (uint64_t)(uintptr_t)kg;
    const uint64_t va = (uint64_t)(uintptr_t)vg;
#pragma unroll
    for (int j = 0; j < 8; ++j) {
        asm volatile("global_load_async_to_lds_b128 %0, %1, off"
                     :: "v"(lbase + (unsigned)j * 16u),
                        "v"(ka + (uint64_t)j * 16u) : "memory");
        asm volatile("global_load_async_to_lds_b128 %0, %1, off"
                     :: "v"(lbase + STAGE_VOFF + (unsigned)j * 16u),
                        "v"(va + (uint64_t)j * 16u) : "memory");
    }
}

__global__ __launch_bounds__(128) void
ScanAttention_fa2_kernel(const float* __restrict__ qs,
                         const float* __restrict__ ks,
                         const float* __restrict__ vs,
                         const int*   __restrict__ valid_lens,
                         float*       __restrict__ out) {
    extern __shared__ char smem[];
    _Float16* Kl = (_Float16*)(smem + KOFF);   // [key][ROWP]
    _Float16* Vl = (_Float16*)(smem + VOFF);   // [d][ROWP]  (transposed)
    _Float16* Pl = (_Float16*)(smem + POFF);   // [wave*16+row][ROWP]

    const int tid  = threadIdx.x;
    const int w    = tid >> 5;          // wave 0..3
    const int lane = tid & 31;
    const int n    = lane & 15;
    const int g    = lane >> 4;

    const int qt = blockIdx.x;
    const int h  = blockIdx.y;
    const int b  = blockIdx.z;

    const int   valid  = valid_lens[b];
    const int   ktiles = (valid + KTILE - 1) / KTILE;
    const float scale  = 0.125f;        // 1/sqrt(64)
    const size_t tokstride = (size_t)HEADS * DHEAD;

    // ---- Q fragments (A layout, scale folded in) ----
    const int qrow = qt * QTILE + w * 16 + n;
    const float* qp = qs + ((size_t)b * QLEN + qrow) * tokstride + (size_t)h * DHEAD;
    v16h a0, a1;
#pragma unroll
    for (int i = 0; i < 8; ++i) {
        a0[i]     = (_Float16)(qp[      g * 8 + i] * scale);
        a0[i + 8] = (_Float16)(qp[16  + g * 8 + i] * scale);
        a1[i]     = (_Float16)(qp[32  + g * 8 + i] * scale);
        a1[i + 8] = (_Float16)(qp[48  + g * 8 + i] * scale);
    }

    float rowm[8], rowl[8];
    v8f acc[4];
#pragma unroll
    for (int r = 0; r < 8; ++r) { rowm[r] = -3.0e38f; rowl[r] = 0.0f; }
#pragma unroll
    for (int nt = 0; nt < 4; ++nt)
#pragma unroll
        for (int r = 0; r < 8; ++r) acc[nt][r] = 0.0f;

    // cooperative loader mapping: 128 threads -> 64 rows x 2 halves of D
    const int crow  = tid >> 1;
    const int chalf = tid & 1;
    const unsigned lstage = ((unsigned)crow * 64u + (unsigned)chalf * 32u) * 4u;

    const float* kg0 = ks + (((size_t)b * KLEN + crow) * HEADS + h) * DHEAD + chalf * 32;
    const float* vg0 = vs + (((size_t)b * KLEN + crow) * HEADS + h) * DHEAD + chalf * 32;
    const size_t tilestep = (size_t)KTILE * tokstride;

    // prologue: stage tile 0 into buffer 0
    issue_tile_async(kg0, vg0, lstage);

    for (int kt = 0; kt < ktiles; ++kt) {
        const unsigned curbase = (kt & 1) ? STAGE_BUF : 0u;
        const unsigned nxtbase = (kt & 1) ? 0u : STAGE_BUF;

        // overlap: stage tile kt+1 while we process tile kt
        if (kt + 1 < ktiles) {
            issue_tile_async(kg0 + (size_t)(kt + 1) * tilestep,
                             vg0 + (size_t)(kt + 1) * tilestep,
                             nxtbase + lstage);
            asm volatile("s_wait_asynccnt 16" ::: "memory");  // tile kt landed (in-order)
        } else {
            asm volatile("s_wait_asynccnt 0" ::: "memory");
        }
        __syncthreads();   // all waves: tile kt staged, previous f16 tiles consumed

        // ---- convert staged f32 -> f16 (K row-major, V transposed) ----
        {
            const float4* sk = (const float4*)(smem + curbase + lstage);
            const float4* sv = (const float4*)(smem + curbase + STAGE_VOFF + lstage);
#pragma unroll
            for (int j = 0; j < 8; ++j) {
                float4 f = sk[j];
                int d0 = chalf * 32 + j * 4;
                Kl[crow * ROWP + d0 + 0] = (_Float16)f.x;
                Kl[crow * ROWP + d0 + 1] = (_Float16)f.y;
                Kl[crow * ROWP + d0 + 2] = (_Float16)f.z;
                Kl[crow * ROWP + d0 + 3] = (_Float16)f.w;
            }
#pragma unroll
            for (int j = 0; j < 8; ++j) {
                float4 f = sv[j];
                int d0 = chalf * 32 + j * 4;
                Vl[(d0 + 0) * ROWP + crow] = (_Float16)f.x;
                Vl[(d0 + 1) * ROWP + crow] = (_Float16)f.y;
                Vl[(d0 + 2) * ROWP + crow] = (_Float16)f.z;
                Vl[(d0 + 3) * ROWP + crow] = (_Float16)f.w;
            }
        }
        __syncthreads();

        // ---- S = Q * K^T ----
        v8f s[4];
#pragma unroll
        for (int nt = 0; nt < 4; ++nt) {
            const v8h* rowp = (const v8h*)(Kl + (nt * 16 + n) * ROWP);
            V16U b0, b1;
            b0.h[0] = rowp[g];     b0.h[1] = rowp[2 + g];
            b1.h[0] = rowp[4 + g]; b1.h[1] = rowp[6 + g];
            v8f c;
#pragma unroll
            for (int r = 0; r < 8; ++r) c[r] = 0.0f;
            c = wmma_f16(a0, b0.v, c);
            c = wmma_f16(a1, b1.v, c);
            s[nt] = c;
        }

        // ---- mask + online softmax (row stats within each 16-lane group) ----
        float tmax[8];
#pragma unroll
        for (int r = 0; r < 8; ++r) tmax[r] = -3.0e38f;
#pragma unroll
        for (int nt = 0; nt < 4; ++nt) {
            const int key = kt * KTILE + nt * 16 + n;
            const bool ok = key < valid;
#pragma unroll
            for (int r = 0; r < 8; ++r) {
                float v = ok ? s[nt][r] : -3.0e38f;
                s[nt][r] = v;
                tmax[r] = fmaxf(tmax[r], v);
            }
        }
#pragma unroll
        for (int off = 1; off < 16; off <<= 1)
#pragma unroll
            for (int r = 0; r < 8; ++r)
                tmax[r] = fmaxf(tmax[r], __shfl_xor(tmax[r], off, 32));

        float sf[8], tsum[8];
#pragma unroll
        for (int r = 0; r < 8; ++r) {
            float mn = fmaxf(rowm[r], tmax[r]);
            sf[r]   = __expf(rowm[r] - mn);
            rowm[r] = mn;
            tsum[r] = 0.0f;
        }
#pragma unroll
        for (int nt = 0; nt < 4; ++nt)
#pragma unroll
            for (int r = 0; r < 8; ++r) {
                float p = __expf(s[nt][r] - rowm[r]);
                s[nt][r] = p;
                tsum[r] += p;
            }
#pragma unroll
        for (int off = 1; off < 16; off <<= 1)
#pragma unroll
            for (int r = 0; r < 8; ++r)
                tsum[r] += __shfl_xor(tsum[r], off, 32);
#pragma unroll
        for (int r = 0; r < 8; ++r) rowl[r] = rowl[r] * sf[r] + tsum[r];
#pragma unroll
        for (int nt = 0; nt < 4; ++nt)
#pragma unroll
            for (int r = 0; r < 8; ++r) acc[nt][r] *= sf[r];

        // ---- P (f16) via per-wave LDS: C layout -> A layout ----
#pragma unroll
        for (int nt = 0; nt < 4; ++nt)
#pragma unroll
            for (int r = 0; r < 8; ++r)
                Pl[(w * 16 + r + 8 * g) * ROWP + nt * 16 + n] = (_Float16)s[nt][r];

        asm volatile("s_wait_dscnt 0" ::: "memory");   // wave-local LDS RAW

        // ---- O += P * V ----
        const v8h* pp = (const v8h*)(Pl + (w * 16 + n) * ROWP);
        V16U pa0, pa1;
        pa0.h[0] = pp[g];     pa0.h[1] = pp[2 + g];
        pa1.h[0] = pp[4 + g]; pa1.h[1] = pp[6 + g];
#pragma unroll
        for (int nt = 0; nt < 4; ++nt) {
            const v8h* vp = (const v8h*)(Vl + (nt * 16 + n) * ROWP);
            V16U vb0, vb1;
            vb0.h[0] = vp[g];     vb0.h[1] = vp[2 + g];
            vb1.h[0] = vp[4 + g]; vb1.h[1] = vp[6 + g];
            acc[nt] = wmma_f16(pa0.v, vb0.v, acc[nt]);
            acc[nt] = wmma_f16(pa1.v, vb1.v, acc[nt]);
        }
    }

    // ---- normalize and write O ----
    float inv[8];
#pragma unroll
    for (int r = 0; r < 8; ++r) inv[r] = 1.0f / rowl[r];
#pragma unroll
    for (int nt = 0; nt < 4; ++nt)
#pragma unroll
        for (int r = 0; r < 8; ++r) {
            const int orow = qt * QTILE + w * 16 + r + 8 * g;
            out[(((size_t)b * QLEN + orow) * HEADS + h) * DHEAD + nt * 16 + n] =
                acc[nt][r] * inv[r];
        }
}

extern "C" void kernel_launch(void* const* d_in, const int* in_sizes, int n_in,
                              void* d_out, int out_size, void* d_ws, size_t ws_size,
                              hipStream_t stream) {
    const float* qs = (const float*)d_in[0];
    const float* ks = (const float*)d_in[1];
    const float* vs = (const float*)d_in[2];
    const int*   vl = (const int*)d_in[3];
    float* out = (float*)d_out;

    dim3 grid(QLEN / QTILE, HEADS, BATCH);
    dim3 block(128);
    ScanAttention_fa2_kernel<<<grid, block, SMEM_TOTAL, stream>>>(qs, ks, vs, vl, out);
}
